// NVFP4_18LayerPersistentMLP_30416958390328
// MI455X (gfx1250) — matrix operand order repaired and checked by reading, not compiled, not run on hardware
//
#include <hip/hip_runtime.h>
#include <hip/hip_bf16.h>

#define NLAYER 18
#define B_ 256
#define H_ 2048
#define F_ 8192
#define QB 32      // quant block size
#define LDK 64     // K chunk staged in LDS
#define LDSS 72    // LDS row stride in f16 (64 + 8 pad, keeps 16B alignment)

typedef __attribute__((ext_vector_type(16))) _Float16 v16h;
typedef __attribute__((ext_vector_type(8)))  _Float16 v8h;
typedef __attribute__((ext_vector_type(8)))  float    v8f;
typedef __attribute__((ext_vector_type(4)))  int      v4i;

// ---------------- CDNA5 async global->LDS copy (ASYNCcnt), guarded fallback ----
#if defined(__has_builtin)
#if __has_builtin(__builtin_amdgcn_global_load_async_to_lds_b128)
#define HAVE_ASYNC_LDS 1
#endif
#endif

#ifdef HAVE_ASYNC_LDS
typedef __attribute__((address_space(1))) v4i as1_v4i;
typedef __attribute__((address_space(3))) v4i as3_v4i;
#endif

__device__ __forceinline__ void cp_async16(void* lds, const void* g) {
#ifdef HAVE_ASYNC_LDS
    __builtin_amdgcn_global_load_async_to_lds_b128(
        (as1_v4i*)g, (as3_v4i*)lds, 0, 0);
#else
    *(v8h*)lds = *(const v8h*)g;
#endif
}

__device__ __forceinline__ void async_wait_all() {
#ifdef HAVE_ASYNC_LDS
#if __has_builtin(__builtin_amdgcn_s_wait_asynccnt)
    __builtin_amdgcn_s_wait_asynccnt(0);
#else
    asm volatile("s_wait_asynccnt 0x0" ::: "memory");
#endif
#endif
}

// ---- NVFP4 decode: idx -> {0,.5,1,1.5,2,3,4,6} with sign, branchless bit build
__device__ __forceinline__ float nvfp4_val(unsigned idx) {
    unsigned m = idx & 7u;
    unsigned s = (idx >> 3) & 1u;
    unsigned bits = ((126u + (m >> 1)) << 23) | ((m & 1u) << 22);
    bits = (m < 2u) ? (m ? 0x3F000000u : 0u) : bits;
    bits |= (s << 31);
    return __uint_as_float(bits);
}

// 4 packed bytes (one per int32 lane of v4i) -> 8 f16 weights (low nibble = even k)
__device__ __forceinline__ v8h dequant8(v4i w, float scale) {
    v8h r;
#pragma unroll
    for (int e = 0; e < 4; ++e) {
        unsigned byte = ((unsigned)w[e]) & 0xFFu;
        float lo = nvfp4_val(byte & 15u) * scale;
        float hi = nvfp4_val((byte >> 4) & 15u) * scale;
        r[2 * e]     = (_Float16)lo;
        r[2 * e + 1] = (_Float16)hi;
    }
    return r;
}

// A fragment (16x32 f16): lanes 0-15 rows, lane-half splits K as interleaved 8-blocks
__device__ __forceinline__ v16h load_fragA(const _Float16* s, int row0, int kk, int lane) {
    int r = row0 + (lane & 15);
    int c = kk + ((lane >> 4) << 3);
    v8h lo = *(const v8h*)(s + r * LDSS + c);
    v8h hi = *(const v8h*)(s + r * LDSS + c + 16);
    return __builtin_shufflevector(lo, hi, 0,1,2,3,4,5,6,7,8,9,10,11,12,13,14,15);
}

// B fragment (32x16 f16): lanes hold N columns, lane-half splits K as interleaved 8-blocks
__device__ __forceinline__ v16h load_fragB(const _Float16* s, int col0, int kk, int lane) {
    int n = col0 + (lane & 15);
    int c = kk + ((lane >> 4) << 3);
    v8h lo = *(const v8h*)(s + n * LDSS + c);
    v8h hi = *(const v8h*)(s + n * LDSS + c + 16);
    return __builtin_shufflevector(lo, hi, 0,1,2,3,4,5,6,7,8,9,10,11,12,13,14,15);
}

__device__ __forceinline__ v8f wmma_f16(v16h a, v16h b, v8f c) {
    return __builtin_amdgcn_wmma_f32_16x16x32_f16(false, a, false, b, (short)0, c, false, false);
}

// ---------------- gate+up fused GEMM: H = silu(A Gw^T) * (A Uw^T) ----------------
__global__ __launch_bounds__(256) void gateup_kernel(
    const _Float16* __restrict__ A,      // [B_, H_] f16
    const int*      __restrict__ gw,     // [F_, H_/2] packed bytes in int32
    const float*    __restrict__ gs,     // [F_, H_/QB]
    const int*      __restrict__ uw,
    const float*    __restrict__ us,
    _Float16*       __restrict__ Hout)   // [B_, F_] f16
{
    __shared__ __align__(16) _Float16 As[2][128 * LDSS];
    __shared__ __align__(16) _Float16 Gs[2][64 * LDSS];
    __shared__ __align__(16) _Float16 Us[2][64 * LDSS];

    const int tid  = threadIdx.x;
    const int lane = tid & 31;
    const int wave = tid >> 5;
    const int wm   = wave >> 1;          // 0..3 -> M sub-tile of 32
    const int wn   = wave & 1;           // 0..1 -> N sub-tile of 32
    const int n0   = blockIdx.x * 64;    // over F_
    const int m0   = blockIdx.y * 128;   // over B_

    const v8f zero = {0.f,0.f,0.f,0.f,0.f,0.f,0.f,0.f};
    v8f accG[2][2], accU[2][2];
#pragma unroll
    for (int i = 0; i < 2; ++i)
#pragma unroll
        for (int j = 0; j < 2; ++j) { accG[i][j] = zero; accU[i][j] = zero; }

    // weight staging registers (loaded early, dequantized late to hide HBM latency)
    v4i wreg_g[2], wreg_u[2];
    float sreg_g[2], sreg_u[2];

    auto stageA = [&](int k0, int buf) {
#pragma unroll
        for (int it = 0; it < 4; ++it) {
            int i = it * 256 + tid;              // 1024 chunks of 8 f16
            int r = i >> 3, c = (i & 7) << 3;
            cp_async16(&As[buf][r * LDSS + c], A + (size_t)(m0 + r) * H_ + k0 + c);
        }
    };
    auto loadW = [&](int k0) {
#pragma unroll
        for (int it = 0; it < 2; ++it) {
            int i = it * 256 + tid;              // 512 int4 chunks per matrix
            int r = i >> 3, j = i & 7;
            int kb = (k0 + j * 8) >> 5;
            wreg_g[it] = *(const v4i*)(gw + (size_t)(n0 + r) * (H_ / 2) + (k0 >> 1) + j * 4);
            sreg_g[it] = gs[(size_t)(n0 + r) * (H_ / QB) + kb];
            wreg_u[it] = *(const v4i*)(uw + (size_t)(n0 + r) * (H_ / 2) + (k0 >> 1) + j * 4);
            sreg_u[it] = us[(size_t)(n0 + r) * (H_ / QB) + kb];
        }
    };
    auto storeW = [&](int buf) {
#pragma unroll
        for (int it = 0; it < 2; ++it) {
            int i = it * 256 + tid;
            int r = i >> 3, j = i & 7;
            *(v8h*)&Gs[buf][r * LDSS + j * 8] = dequant8(wreg_g[it], sreg_g[it]);
            *(v8h*)&Us[buf][r * LDSS + j * 8] = dequant8(wreg_u[it], sreg_u[it]);
        }
    };

    const int NC = H_ / LDK;
    stageA(0, 0);
    loadW(0);
    storeW(0);

    for (int c = 0; c < NC; ++c) {
        const int buf = c & 1;
        async_wait_all();
        __syncthreads();

        if (c + 1 < NC) {                 // issue next-chunk loads under the WMMAs
            stageA((c + 1) * LDK, buf ^ 1);
            loadW((c + 1) * LDK);
        }
        if (c + 2 < NC) {                 // global_prefetch_b8 two chunks ahead
            int r = tid >> 2, j = tid & 3;
            int kp = (c + 2) * LDK;
            __builtin_prefetch(gw + (size_t)(n0 + r) * (H_ / 2) + (kp >> 1) + j * 8, 0, 1);
            __builtin_prefetch(uw + (size_t)(n0 + r) * (H_ / 2) + (kp >> 1) + j * 8, 0, 1);
        }

        const _Float16* Ab = &As[buf][0];
        const _Float16* Gb = &Gs[buf][0];
        const _Float16* Ub = &Us[buf][0];
#pragma unroll
        for (int kk = 0; kk < LDK; kk += 32) {
            v16h a0 = load_fragA(Ab, wm * 32 + 0,  kk, lane);
            v16h a1 = load_fragA(Ab, wm * 32 + 16, kk, lane);
#pragma unroll
            for (int ni = 0; ni < 2; ++ni) {
                v16h bg = load_fragB(Gb, wn * 32 + ni * 16, kk, lane);
                v16h bu = load_fragB(Ub, wn * 32 + ni * 16, kk, lane);
                accG[0][ni] = wmma_f16(a0, bg, accG[0][ni]);
                accG[1][ni] = wmma_f16(a1, bg, accG[1][ni]);
                accU[0][ni] = wmma_f16(a0, bu, accU[0][ni]);
                accU[1][ni] = wmma_f16(a1, bu, accU[1][ni]);
            }
        }
        if (c + 1 < NC) storeW(buf ^ 1);  // dequant after compute; barrier next iter
    }

    // epilogue: silu(gate) * up -> f16. D layout: m = e + 8*(lane>=16), n = lane&15
#pragma unroll
    for (int mi = 0; mi < 2; ++mi)
#pragma unroll
        for (int ni = 0; ni < 2; ++ni)
#pragma unroll
            for (int e = 0; e < 8; ++e) {
                float g = accG[mi][ni][e];
                float u = accU[mi][ni][e];
                float h = (g / (1.0f + __expf(-g))) * u;
                int m = m0 + wm * 32 + mi * 16 + e + ((lane >> 4) << 3);
                int n = n0 + wn * 32 + ni * 16 + (lane & 15);
                Hout[(size_t)m * F_ + n] = (_Float16)h;
            }
}

// ---------------- down GEMM (split-K): P[z] = Hin[:, kz] Dw[:, kz]^T --------------
__global__ __launch_bounds__(256) void down_kernel(
    const _Float16* __restrict__ Hin,    // [B_, F_] f16
    const int*      __restrict__ dw,     // [H_, F_/2]
    const float*    __restrict__ dsc,    // [H_, F_/QB]
    float*          __restrict__ P,      // [splits, B_, H_] f32 partials
    int kspan)                           // K per split
{
    __shared__ __align__(16) _Float16 As[2][128 * LDSS];
    __shared__ __align__(16) _Float16 Ws[2][64 * LDSS];

    const int tid  = threadIdx.x;
    const int lane = tid & 31;
    const int wave = tid >> 5;
    const int wm   = wave >> 1;
    const int wn   = wave & 1;
    const int n0   = blockIdx.x * 64;    // over H_
    const int m0   = blockIdx.y * 128;   // over B_
    const int kb0  = blockIdx.z * kspan; // K split origin
    float* Pz = P + (size_t)blockIdx.z * B_ * H_;

    const v8f zero = {0.f,0.f,0.f,0.f,0.f,0.f,0.f,0.f};
    v8f acc[2][2];
#pragma unroll
    for (int i = 0; i < 2; ++i)
#pragma unroll
        for (int j = 0; j < 2; ++j) acc[i][j] = zero;

    v4i wreg[2];
    float sreg[2];

    auto stageA = [&](int k0, int buf) {
#pragma unroll
        for (int it = 0; it < 4; ++it) {
            int i = it * 256 + tid;
            int r = i >> 3, c = (i & 7) << 3;
            cp_async16(&As[buf][r * LDSS + c], Hin + (size_t)(m0 + r) * F_ + k0 + c);
        }
    };
    auto loadW = [&](int k0) {
#pragma unroll
        for (int it = 0; it < 2; ++it) {
            int i = it * 256 + tid;
            int r = i >> 3, j = i & 7;
            int kb = (k0 + j * 8) >> 5;
            wreg[it] = *(const v4i*)(dw + (size_t)(n0 + r) * (F_ / 2) + (k0 >> 1) + j * 4);
            sreg[it] = dsc[(size_t)(n0 + r) * (F_ / QB) + kb];
        }
    };
    auto storeW = [&](int buf) {
#pragma unroll
        for (int it = 0; it < 2; ++it) {
            int i = it * 256 + tid;
            int r = i >> 3, j = i & 7;
            *(v8h*)&Ws[buf][r * LDSS + j * 8] = dequant8(wreg[it], sreg[it]);
        }
    };

    const int NC = kspan / LDK;
    stageA(kb0, 0);
    loadW(kb0);
    storeW(0);

    for (int c = 0; c < NC; ++c) {
        const int buf = c & 1;
        async_wait_all();
        __syncthreads();

        if (c + 1 < NC) {
            stageA(kb0 + (c + 1) * LDK, buf ^ 1);
            loadW(kb0 + (c + 1) * LDK);
        }
        if (c + 2 < NC) {
            int r = tid >> 2, j = tid & 3;
            int kp = kb0 + (c + 2) * LDK;
            __builtin_prefetch(dw + (size_t)(n0 + r) * (F_ / 2) + (kp >> 1) + j * 8, 0, 1);
        }

        const _Float16* Ab = &As[buf][0];
        const _Float16* Wb = &Ws[buf][0];
#pragma unroll
        for (int kk = 0; kk < LDK; kk += 32) {
            v16h a0 = load_fragA(Ab, wm * 32 + 0,  kk, lane);
            v16h a1 = load_fragA(Ab, wm * 32 + 16, kk, lane);
#pragma unroll
            for (int ni = 0; ni < 2; ++ni) {
                v16h b = load_fragB(Wb, wn * 32 + ni * 16, kk, lane);
                acc[0][ni] = wmma_f16(a0, b, acc[0][ni]);
                acc[1][ni] = wmma_f16(a1, b, acc[1][ni]);
            }
        }
        if (c + 1 < NC) storeW(buf ^ 1);
    }

#pragma unroll
    for (int mi = 0; mi < 2; ++mi)
#pragma unroll
        for (int ni = 0; ni < 2; ++ni)
#pragma unroll
            for (int e = 0; e < 8; ++e) {
                int m = m0 + wm * 32 + mi * 16 + e + ((lane >> 4) << 3);
                int n = n0 + wn * 32 + ni * 16 + (lane & 15);
                Pz[(size_t)m * H_ + n] = acc[mi][ni][e];
            }
}

// ---------------- split-K reduce + next-layer activation convert -----------------
__global__ void reduce_kernel(const float* __restrict__ P, int splits,
                              _Float16* __restrict__ Aout, float* __restrict__ OutF32) {
    int i = blockIdx.x * blockDim.x + threadIdx.x;   // B_*H_ elements
    float s = 0.f;
    for (int z = 0; z < splits; ++z) s += P[(size_t)z * B_ * H_ + i];
    Aout[i] = (_Float16)s;
    if (OutF32) OutF32[i] = s;
}

__global__ void cvt_f32_to_f16(const float* __restrict__ x, _Float16* __restrict__ y, int n) {
    int i = blockIdx.x * blockDim.x + threadIdx.x;
    if (i < n) y[i] = (_Float16)x[i];
}

extern "C" void kernel_launch(void* const* d_in, const int* in_sizes, int n_in,
                              void* d_out, int out_size, void* d_ws, size_t ws_size,
                              hipStream_t stream) {
    (void)in_sizes; (void)n_in; (void)out_size;
    const float* x   = (const float*)d_in[0];
    const int*   gp  = (const int*)  d_in[1];
    const float* gsc = (const float*)d_in[2];
    const int*   upk = (const int*)  d_in[3];
    const float* usc = (const float*)d_in[4];
    const int*   dpk = (const int*)  d_in[5];
    const float* dsc = (const float*)d_in[6];
    float* out = (float*)d_out;

    _Float16* Af16 = (_Float16*)d_ws;                               // 1 MB @ 0
    _Float16* Hf16 = (_Float16*)((char*)d_ws + (size_t)(2u << 20)); // 4 MB @ 2 MB
    float*    P    = (float*)   ((char*)d_ws + (size_t)(6u << 20)); // partials @ 6 MB

    // deterministic split-K selection from workspace size (2 MB per split)
    const int splits = (ws_size >= (size_t)(14u << 20)) ? 4 : 1;
    const int kspan  = F_ / splits;

    cvt_f32_to_f16<<<(B_ * H_ + 255) / 256, 256, 0, stream>>>(x, Af16, B_ * H_);

    dim3 blk(256);
    dim3 g1(F_ / 64, B_ / 128);            // gate/up: 256 WGs
    dim3 g2(H_ / 64, B_ / 128, splits);    // down:    64*splits WGs
    dim3 gr((B_ * H_) / 256);

    const size_t wstride_gu = (size_t)F_ * (H_ / 2);
    const size_t sstride_gu = (size_t)F_ * (H_ / QB);
    const size_t wstride_d  = (size_t)H_ * (F_ / 2);
    const size_t sstride_d  = (size_t)H_ * (F_ / QB);

    for (int l = 0; l < NLAYER; ++l) {
        gateup_kernel<<<g1, blk, 0, stream>>>(
            Af16,
            gp + (size_t)l * wstride_gu, gsc + (size_t)l * sstride_gu,
            upk + (size_t)l * wstride_gu, usc + (size_t)l * sstride_gu,
            Hf16);
        down_kernel<<<g2, blk, 0, stream>>>(
            Hf16,
            dpk + (size_t)l * wstride_d, dsc + (size_t)l * sstride_d,
            P, kspan);
        reduce_kernel<<<gr, blk, 0, stream>>>(
            P, splits, Af16, (l == NLAYER - 1) ? out : (float*)nullptr);
    }
}
